// AttentionMobile_8804682957364
// MI455X (gfx1250) — matrix operand order repaired and checked by reading, not compiled, not run on hardware
//
#include <hip/hip_runtime.h>

// ---------------------------------------------------------------------------
// CDNA5 (gfx1250) ReLU-attention:
//   - all GEMM stages on v_wmma_f32_16x16x32_bf16 (f32 accum)
//   - tile staging via GLOBAL_LOAD_ASYNC_TO_LDS_B128 (+ s_wait_asynccnt),
//     double-buffered so the async DMA overlaps the WMMA chain
// ---------------------------------------------------------------------------

typedef __attribute__((ext_vector_type(16))) __bf16 v16bf;
typedef __attribute__((ext_vector_type(8)))  float  v8f;

struct __attribute__((aligned(16))) U4 { unsigned int x, y, z, w; };
union Frag16 { v16bf v; U4 u[2]; };

static __device__ __forceinline__ unsigned short f2bf(float f) {
    unsigned int u = __float_as_uint(f);
    u += 0x7fffu + ((u >> 16) & 1u);          // round-to-nearest-even
    return (unsigned short)(u >> 16);
}

static __device__ __forceinline__ U4 ldg4(const unsigned short* p) {
    return *reinterpret_cast<const U4*>(p);
}
static __device__ __forceinline__ U4 lds4(const unsigned short* p) {
    return *reinterpret_cast<const U4*>(p);
}

// Async global -> LDS copy of 16 bytes, per lane. The low 32 bits of a
// generic pointer to __shared__ memory ARE the LDS byte offset on AMDGPU.
static __device__ __forceinline__ void async_ld_b128(unsigned short* lds_p,
                                                     const unsigned short* gp) {
    unsigned loff = (unsigned)(unsigned long long)(void*)lds_p;
    asm volatile("global_load_async_to_lds_b128 %0, %1, off"
                 :: "v"(loff), "v"(gp)
                 : "memory");
}
static __device__ __forceinline__ void wait_async0() {
    asm volatile("s_wait_asynccnt 0x0" ::: "memory");
}

// ---------------------------------------------------------------------------
// fp32 -> bf16 elementwise convert (4 elems / thread)
// ---------------------------------------------------------------------------
__global__ __launch_bounds__(256) void cvt_f32_to_bf16(
    const float* __restrict__ in, unsigned short* __restrict__ out, int n4)
{
    int i = blockIdx.x * blockDim.x + threadIdx.x;
    if (i >= n4) return;
    float4 f = reinterpret_cast<const float4*>(in)[i];
    struct __attribute__((aligned(8))) US4 { unsigned short a, b, c, d; } o;
    o.a = f2bf(f.x); o.b = f2bf(f.y); o.c = f2bf(f.z); o.d = f2bf(f.w);
    *reinterpret_cast<US4*>(out + (size_t)i * 4) = o;
}

// ---------------------------------------------------------------------------
// W [K][N] f32  ->  Wt [N][K] bf16   (32x32 LDS tile transpose)
// ---------------------------------------------------------------------------
__global__ __launch_bounds__(256) void transpose_cvt(
    const float* __restrict__ W, unsigned short* __restrict__ Wt, int K, int N)
{
    __shared__ float tile[32][33];
    const int bn = blockIdx.x * 32;          // N offset
    const int bk = blockIdx.y * 32;          // K offset
    const int tx = threadIdx.x & 31;
    const int ty = threadIdx.x >> 5;         // 0..7
#pragma unroll
    for (int r = 0; r < 32; r += 8)
        tile[ty + r][tx] = W[(size_t)(bk + ty + r) * N + bn + tx];
    __syncthreads();
#pragma unroll
    for (int r = 0; r < 32; r += 8)
        Wt[(size_t)(bn + ty + r) * K + bk + tx] = f2bf(tile[tx][ty + r]);
}

// ---------------------------------------------------------------------------
// kvb [B*S][1024] bf16 (head h at cols h*64..)  ->  kvT [B*H][64][S] bf16
// ---------------------------------------------------------------------------
__global__ __launch_bounds__(256) void transpose_kv(
    const unsigned short* __restrict__ kvb, unsigned short* __restrict__ kvT,
    int S)
{
    __shared__ unsigned short t32[32][33];
    const int s0 = blockIdx.x * 32;
    const int d0 = blockIdx.y * 32;
    const int bh = blockIdx.z;
    const int b = bh >> 4, h = bh & 15;
    const int tx = threadIdx.x & 31;
    const int ty = threadIdx.x >> 5;
#pragma unroll
    for (int r = 0; r < 32; r += 8)
        t32[ty + r][tx] =
            kvb[(size_t)(b * S + s0 + ty + r) * 1024 + h * 64 + d0 + tx];
    __syncthreads();
#pragma unroll
    for (int r = 0; r < 32; r += 8)
        kvT[((size_t)bh * 64 + d0 + ty + r) * S + s0 + tx] = t32[tx][ty + r];
}

// ---------------------------------------------------------------------------
// C[M][N] = A[M][K] @ Bt[N][K]^T  (bf16 in, f32 accum).
// Block: 8 waves as 4(M) x 2(N); block tile 128x128; wave tile 32x64.
// K stepped by 32; double-buffered async LDS staging pipelined vs WMMA.
// ---------------------------------------------------------------------------
template <bool F32OUT>
__global__ __launch_bounds__(256) void gemm_bf16(
    const unsigned short* __restrict__ A,   // [M][K] bf16 row-major
    const unsigned short* __restrict__ Bt,  // [N][K] bf16 row-major
    const float* __restrict__ bias,         // [N] (used when F32OUT)
    unsigned short* __restrict__ Cb,        // bf16 out (used when !F32OUT)
    float* __restrict__ Cf,                 // f32 out  (used when F32OUT)
    int M, int N, int K)
{
    __shared__ unsigned short sA[2][128 * 32];
    __shared__ unsigned short sB[2][128 * 32];

    const int t    = threadIdx.x;
    const int lane = t & 31;
    const int w    = t >> 5;        // wave 0..7
    const int wm   = w >> 1;        // 0..3 (M)
    const int wn   = w & 1;         // 0..1 (N)
    const int half = lane >> 4;     // 0/1
    const int l16  = lane & 15;
    const int rowBase = blockIdx.y * 128;
    const int colBase = blockIdx.x * 128;

    // staging chunk coordinates (each thread: chunks t and t+256 of 512)
    const int r0 = t >> 2,         c0 = (t & 3) * 8;
    const int r1 = (t + 256) >> 2, c1 = c0;           // (t+256)&3 == t&3

    auto stage = [&](int kk, int buf) {
        async_ld_b128(&sA[buf][r0 * 32 + c0], &A[(size_t)(rowBase + r0) * K + kk + c0]);
        async_ld_b128(&sA[buf][r1 * 32 + c1], &A[(size_t)(rowBase + r1) * K + kk + c1]);
        async_ld_b128(&sB[buf][r0 * 32 + c0], &Bt[(size_t)(colBase + r0) * K + kk + c0]);
        async_ld_b128(&sB[buf][r1 * 32 + c1], &Bt[(size_t)(colBase + r1) * K + kk + c1]);
    };

    const v8f vzero = {0.f, 0.f, 0.f, 0.f, 0.f, 0.f, 0.f, 0.f};
    v8f acc[2][4];
#pragma unroll
    for (int i = 0; i < 2; ++i)
#pragma unroll
        for (int j = 0; j < 4; ++j) acc[i][j] = vzero;

    stage(0, 0);                                  // prefetch first tile
    int cur = 0;
    for (int kk = 0; kk < K; kk += 32) {
        wait_async0();                            // tile kk landed in LDS
        __syncthreads();                          // all waves: LDS ready,
                                                  // other buffer free
        int knext = (kk + 32 < K) ? kk + 32 : 0;  // wrap: branch-free tail
        stage(knext, cur ^ 1);                    // overlaps compute below

        // fragments (ISA 16-bit A layout; B lane-striped by column)
        Frag16 af[2], bf[4];
#pragma unroll
        for (int i = 0; i < 2; ++i) {
            const unsigned short* ar = &sA[cur][(wm * 32 + i * 16 + l16) * 32];
            af[i].u[0] = lds4(ar + half * 8);
            af[i].u[1] = lds4(ar + 16 + half * 8);
        }
#pragma unroll
        for (int j = 0; j < 4; ++j) {
            const unsigned short* br =
                &sB[cur][(wn * 64 + j * 16 + l16) * 32 + half * 8];
            bf[j].u[0] = lds4(br);
            bf[j].u[1] = lds4(br + 16);
        }
#pragma unroll
        for (int i = 0; i < 2; ++i)
#pragma unroll
            for (int j = 0; j < 4; ++j)
                acc[i][j] = __builtin_amdgcn_wmma_f32_16x16x32_bf16(
                    false, af[i].v, false, bf[j].v, (short)0, acc[i][j], false, false);
        cur ^= 1;
    }

    // epilogue: D layout m = r + 8*half, n = l16
#pragma unroll
    for (int i = 0; i < 2; ++i) {
#pragma unroll
        for (int j = 0; j < 4; ++j) {
            const int n = colBase + wn * 64 + j * 16 + l16;
            float bv = 0.f;
            if (F32OUT) bv = bias[n];
#pragma unroll
            for (int r = 0; r < 8; ++r) {
                int m = rowBase + wm * 32 + i * 16 + r + half * 8;
                float v = acc[i][j][r];
                if (F32OUT) Cf[(size_t)m * N + n] = v + bv;
                else        Cb[(size_t)m * N + n] = f2bf(v);
            }
        }
    }
}

// ---------------------------------------------------------------------------
// ReLU attention per (b,h): O = relu(Q K^T * 0.125) V, V == K.
// Block: 128 queries (8 waves x 16). 32-key tiles double-buffered via async
// LDS copies: kd[key][dh] from kvb (QK^T B-frags), dk[dh][key] from kvT.
// ---------------------------------------------------------------------------
__global__ __launch_bounds__(256) void attn_relu(
    const unsigned short* __restrict__ Qb,   // [B*S][1024]
    const unsigned short* __restrict__ KVb,  // [B*S][1024]
    const unsigned short* __restrict__ KVt,  // [B*H][64][S]
    unsigned short* __restrict__ Ctx,        // [B*S][1024]
    int S)
{
    __shared__ unsigned short kd[2][32 * 64];     // [key][dh]
    __shared__ unsigned short dk[2][64 * 32];     // [dh][key]
    __shared__ unsigned short Pt[8][16 * 32];     // per-wave P tile (bf16)

    const int t    = threadIdx.x;
    const int lane = t & 31;
    const int w    = t >> 5;
    const int half = lane >> 4;
    const int l16  = lane & 15;
    const int bh = blockIdx.y;
    const int b  = bh >> 4;
    const int h  = bh & 15;
    const int q0 = blockIdx.x * 128;
    const size_t rs      = 1024;
    const size_t headoff = (size_t)h * 64;
    const size_t baseBS  = (size_t)b * (size_t)S * rs;
    const unsigned short* kvtHead = KVt + (size_t)bh * 64 * S;
    const float  scale   = 0.125f;                // 1/sqrt(64)

    // staging coordinates (one B128 chunk per thread per tile)
    const int kdKey = t >> 3, kdC = (t & 7) * 8;  // kd: 32 rows x 8 chunks
    const int dkDh  = t >> 2, dkC = (t & 3) * 8;  // dk: 64 rows x 4 chunks

    auto stage = [&](int kt, int buf) {
        async_ld_b128(&kd[buf][kdKey * 64 + kdC],
                      KVb + baseBS + (size_t)(kt + kdKey) * rs + headoff + kdC);
        async_ld_b128(&dk[buf][dkDh * 32 + dkC],
                      kvtHead + (size_t)dkDh * S + kt + dkC);
    };

    // Q fragments (A-matrix, 16x32 over dh, two K-chunks): registers
    Frag16 qf[2];
    {
        const unsigned short* qrow =
            Qb + baseBS + (size_t)(q0 + w * 16 + l16) * rs + headoff;
#pragma unroll
        for (int c = 0; c < 2; ++c) {
            qf[c].u[0] = ldg4(qrow + c * 32 + half * 8);
            qf[c].u[1] = ldg4(qrow + c * 32 + 16 + half * 8);
        }
    }

    const v8f vzero = {0.f, 0.f, 0.f, 0.f, 0.f, 0.f, 0.f, 0.f};
    v8f o[4];
#pragma unroll
    for (int j = 0; j < 4; ++j) o[j] = vzero;

    stage(0, 0);
    int cur = 0;
    for (int kt = 0; kt < S; kt += 32) {
        wait_async0();
        __syncthreads();
        int ktn = (kt + 32 < S) ? kt + 32 : 0;    // wrap: branch-free tail
        stage(ktn, cur ^ 1);                      // overlaps compute below

        // scores S = Q K^T for two 16-key groups, then relu*scale -> P tile
#pragma unroll
        for (int sj = 0; sj < 2; ++sj) {
            v8f sacc = vzero;
#pragma unroll
            for (int c = 0; c < 2; ++c) {
                Frag16 bfr;
                const unsigned short* krow =
                    &kd[cur][(sj * 16 + l16) * 64 + c * 32 + half * 8];
                bfr.u[0] = lds4(krow);
                bfr.u[1] = lds4(krow + 16);
                sacc = __builtin_amdgcn_wmma_f32_16x16x32_bf16(
                    false, qf[c].v, false, bfr.v, (short)0, sacc, false, false);
            }
#pragma unroll
            for (int r = 0; r < 8; ++r) {
                float pv = sacc[r] * scale;
                pv = pv > 0.f ? pv : 0.f;          // ReLU (no softmax)
                Pt[w][(r + half * 8) * 32 + sj * 16 + l16] = f2bf(pv);
            }
        }
        // O += P @ V  (P per-wave private: intra-wave DS ordering, no barrier)
        Frag16 pf, vf[4];
        pf.u[0] = lds4(&Pt[w][l16 * 32 + half * 8]);
        pf.u[1] = lds4(&Pt[w][l16 * 32 + 16 + half * 8]);
#pragma unroll
        for (int j = 0; j < 4; ++j) {
            const unsigned short* vrow =
                &dk[cur][(j * 16 + l16) * 32 + half * 8];
            vf[j].u[0] = lds4(vrow);
            vf[j].u[1] = lds4(vrow + 16);
        }
#pragma unroll
        for (int j = 0; j < 4; ++j)
            o[j] = __builtin_amdgcn_wmma_f32_16x16x32_bf16(
                false, pf.v, false, vf[j].v, (short)0, o[j], false, false);
        cur ^= 1;
    }

    // write context (bf16), head-interleaved layout for the final projection
#pragma unroll
    for (int j = 0; j < 4; ++j) {
#pragma unroll
        for (int r = 0; r < 8; ++r) {
            int m  = q0 + w * 16 + r + half * 8;
            int dh = j * 16 + l16;
            Ctx[baseBS + (size_t)m * rs + headoff + dh] = f2bf(o[j][r]);
        }
    }
}

// ---------------------------------------------------------------------------
// launch
// ---------------------------------------------------------------------------
extern "C" void kernel_launch(void* const* d_in, const int* in_sizes, int n_in,
                              void* d_out, int out_size, void* d_ws, size_t ws_size,
                              hipStream_t stream) {
    (void)in_sizes; (void)n_in; (void)out_size; (void)ws_size;
    const float* hidden = (const float*)d_in[0];
    const float* Wq     = (const float*)d_in[1];
    const float* Wkv    = (const float*)d_in[2];
    const float* Wout   = (const float*)d_in[3];
    const float* bout   = (const float*)d_in[4];
    float* out = (float*)d_out;

    const int B = 2, S = 2048, D = 1024, H = 16;
    const int M = B * S;                       // 4096 token rows

    unsigned char* ws = (unsigned char*)d_ws;
    size_t off = 0;
    auto wsalloc = [&](size_t bytes) -> void* {
        void* p = ws + off;
        off += (bytes + 255) & ~(size_t)255;
        return p;
    };
    unsigned short* hb    = (unsigned short*)wsalloc((size_t)M * D * 2);
    unsigned short* wqT   = (unsigned short*)wsalloc((size_t)D * D * 2);
    unsigned short* wkvT  = (unsigned short*)wsalloc((size_t)D * D * 2);
    unsigned short* woutT = (unsigned short*)wsalloc((size_t)D * D * 2);
    unsigned short* qb    = (unsigned short*)wsalloc((size_t)M * D * 2);
    unsigned short* kvb   = (unsigned short*)wsalloc((size_t)M * D * 2);
    unsigned short* kvT   = (unsigned short*)wsalloc((size_t)M * D * 2);
    unsigned short* ctx   = (unsigned short*)wsalloc((size_t)M * D * 2);

    // 1) conversions / weight transposes
    {
        int n4 = (M * D) / 4;
        cvt_f32_to_bf16<<<(n4 + 255) / 256, 256, 0, stream>>>(hidden, hb, n4);
    }
    dim3 tg(D / 32, D / 32);
    transpose_cvt<<<tg, 256, 0, stream>>>(Wq,   wqT,   D, D);
    transpose_cvt<<<tg, 256, 0, stream>>>(Wkv,  wkvT,  D, D);
    transpose_cvt<<<tg, 256, 0, stream>>>(Wout, woutT, D, D);

    // 2) q / kv projections (bf16 out)
    dim3 gg(D / 128, M / 128);
    gemm_bf16<false><<<gg, 256, 0, stream>>>(hb, wqT,  nullptr, qb,  nullptr, M, D, D);
    gemm_bf16<false><<<gg, 256, 0, stream>>>(hb, wkvT, nullptr, kvb, nullptr, M, D, D);

    // 3) per-head transpose of kv for the PV GEMM B-fragments
    transpose_kv<<<dim3(S / 32, 64 / 32, B * H), 256, 0, stream>>>(kvb, kvT, S);

    // 4) ReLU attention (bf16 context out)
    attn_relu<<<dim3(S / 128, B * H), 256, 0, stream>>>(qb, kvb, kvT, ctx, S);

    // 5) output projection (f32 + bias)
    gemm_bf16<true><<<gg, 256, 0, stream>>>(ctx, woutT, bout, nullptr, out, M, D, D);
}
